// TaskAlignedAssigner_9242769621616
// MI455X (gfx1250) — matrix operand order repaired and checked by reading, not compiled, not run on hardware
//
#include <hip/hip_runtime.h>

// ---------------- problem constants ----------------
#define TOPK_N 13
#define NC_N   80
#define BS_N   16
#define NMAX_N 64
#define A_N    8400     // 80^2 + 40^2 + 20^2 anchors
#define ATILES 33       // ceil(8400/256)
#define WTILES 525      // 8400/16 WMMA anchor tiles

typedef __attribute__((ext_vector_type(16))) _Float16 v16h;
typedef __attribute__((ext_vector_type(8)))  float    v8f;

#define GEPS 1e-9f                              // global EPS
#define PI_F 3.14159265358979323846f

// ---------------- CDNA5 async Global->LDS copy (ASYNCcnt path) ----------------
// CDNA5 ISA: GLOBAL_LOAD_ASYNC_TO_LDS_B128, GV mode (saddr=off).
// VDST VGPR carries the 32-bit LDS byte offset (flat LDS aperture truncation).
__device__ __forceinline__ void async_copy_b128(void* lds_dst, const void* gsrc) {
  unsigned loff = (unsigned)(size_t)lds_dst;            // low 32 bits = LDS offset
  unsigned long long ga = (unsigned long long)(size_t)gsrc;
  asm volatile("global_load_async_to_lds_b128 %0, %1, off"
               :: "v"(loff), "v"(ga) : "memory");
}
__device__ __forceinline__ void async_wait0() {
  asm volatile("s_wait_asynccnt 0" ::: "memory");
}

// ---------------- K0: zero the per-gt max accumulators ----------------
__global__ void tal_k0_init(float* __restrict__ pal, float* __restrict__ pov) {
  for (int i = threadIdx.x; i < BS_N * NMAX_N; i += 256) { pal[i] = 0.f; pov[i] = 0.f; }
}

// ---------------- K1: CIoU overlaps (clamped >= 0) ----------------
__global__ void tal_k1_ciou(const float* __restrict__ pd_bboxes,
                            const float* __restrict__ gt_bboxes,
                            float* __restrict__ ov) {
  const int b   = blockIdx.y;
  const int tid = threadIdx.x;
  const int a   = blockIdx.x * 256 + tid;
  __shared__ __align__(16) float4 gboxS[NMAX_N];
  __shared__ float gx1[NMAX_N], gy1[NMAX_N], gx2[NMAX_N], gy2[NMAX_N];
  __shared__ float gat[NMAX_N], gar[NMAX_N], gsx[NMAX_N], gsy[NMAX_N];
  // async-stage the 64 gt boxes (1 KB) straight into LDS
  if (tid < NMAX_N) {
    async_copy_b128(&gboxS[tid],
                    reinterpret_cast<const float4*>(gt_bboxes) + b * NMAX_N + tid);
    async_wait0();
    const float4 bx = gboxS[tid];
    float w1 = bx.z - bx.x;
    float h1 = bx.w - bx.y + 1e-7f;
    gx1[tid] = bx.x; gy1[tid] = bx.y; gx2[tid] = bx.z; gy2[tid] = bx.w;
    gat[tid] = atanf(w1 / h1);
    gar[tid] = w1 * h1;
    gsx[tid] = bx.x + bx.z; gsy[tid] = bx.y + bx.w;
  }
  if (a < A_N) __builtin_prefetch(&pd_bboxes[((size_t)b * A_N + a) * 4], 0, 0);
  __syncthreads();
  if (a >= A_N) return;
  const float4 pb = reinterpret_cast<const float4*>(pd_bboxes)[(size_t)b * A_N + a];
  const float w2  = pb.z - pb.x;
  const float h2  = pb.w - pb.y + 1e-7f;
  const float at2 = atanf(w2 / h2);
  const float ar2 = w2 * h2;
  const float psx = pb.x + pb.z, psy = pb.y + pb.w;
  float* orow = ov + (size_t)b * NMAX_N * A_N + a;
  for (int g = 0; g < NMAX_N; ++g) {
    float iw    = fminf(gx2[g], pb.z) - fmaxf(gx1[g], pb.x);
    float ih    = fminf(gy2[g], pb.w) - fmaxf(gy1[g], pb.y);
    float inter = fmaxf(iw, 0.f) * fmaxf(ih, 0.f);
    float uni   = gar[g] + ar2 - inter + 1e-7f;
    float iou   = inter / uni;
    float cw    = fmaxf(gx2[g], pb.z) - fminf(gx1[g], pb.x);
    float ch    = fmaxf(gy2[g], pb.w) - fminf(gy1[g], pb.y);
    float c2    = cw * cw + ch * ch + 1e-7f;
    float dx    = psx - gsx[g], dy = psy - gsy[g];
    float rho2  = (dx * dx + dy * dy) * 0.25f;
    float dat   = at2 - gat[g];
    float v     = (4.0f / (PI_F * PI_F)) * dat * dat;
    float alpha = v / (v - iou + 1.0f + 1e-7f);
    float ciou  = iou - (rho2 / c2 + v * alpha);
    orow[(size_t)g * A_N] = fmaxf(ciou, 0.f);
  }
}

// ---------------- K2: align_raw = score * ov^6 via one-hot WMMA GEMM ----------------
// scores(b, 16-anchor tile, 64 gts) = S(16xNC) @ onehot(NCx64), K padded 80->96.
__global__ void tal_k2_wmma(const float* __restrict__ pd_scores,
                            const int*   __restrict__ gt_labels,
                            const float* __restrict__ ov,
                            float* __restrict__ alig) {
  const int tile = blockIdx.x;        // 0..524
  const int b    = blockIdx.y;        // 0..15
  const int lane = threadIdx.x;       // 0..31 (one wave32)
  __shared__ int lbl[NMAX_N];
  lbl[lane]      = gt_labels[b * NMAX_N + lane];
  lbl[lane + 32] = gt_labels[b * NMAX_N + lane + 32];
  __syncthreads();
  const int m  = lane & 15;           // A row / D col-within-tile
  const int hi = lane >> 4;           // lane group
  const int a  = tile * 16 + m;
  const float* srow = pd_scores + ((size_t)b * A_N + a) * NC_N;

  // A-matrix 16x32 f16 layout (ISA 7.12.2): low lanes K=0..7,16..23; high lanes +8
  v16h Ak[3];
  #pragma unroll
  for (int kb = 0; kb < 3; ++kb) {
    const int kOff = kb * 32 + hi * 8;
    v16h av;
    #pragma unroll
    for (int e = 0; e < 16; ++e) {
      int k = kOff + (e < 8 ? e : e + 8);
      float s = (k < NC_N) ? srow[k] : 0.0f;
      av[e] = (_Float16)s;
    }
    Ak[kb] = av;
  }

  #pragma unroll
  for (int nt = 0; nt < 4; ++nt) {
    const int n = nt * 16 + m;        // gt column
    const int L = lbl[n];
    v8f c = {};
    #pragma unroll
    for (int kb = 0; kb < 3; ++kb) {
      // B-matrix 32x16: lanes 0-15 K=0..7,16..23, lanes 16-31 +8, col = lane&15
      const int kOff = kb * 32 + hi * 8;
      v16h bv;
      #pragma unroll
      for (int e = 0; e < 16; ++e) {
        int k = kOff + (e < 8 ? e : e + 8);
        bv[e] = (L == k) ? (_Float16)1.0f : (_Float16)0.0f;
      }
      c = __builtin_amdgcn_wmma_f32_16x16x32_f16(false, Ak[kb], false, bv,
                                                 (short)0, c, false, false);
    }
    // D 16x16 f32: VGPR r -> M = r + hi*8 (anchor), lane&15 -> N (gt)
    #pragma unroll
    for (int r = 0; r < 8; ++r) {
      const int mm = r + hi * 8;
      const int aa = tile * 16 + mm;
      const int g  = nt * 16 + m;
      const size_t idx = ((size_t)b * NMAX_N + g) * A_N + aa;
      float o  = ov[idx];
      float o2 = o * o;
      alig[idx] = c[r] * (o2 * o2 * o2);   // score^ALPHA * ov^BETA, ALPHA=1 BETA=6
    }
  }
}

// ---------------- K3: per-(b,gt) top-13 of align * mask_in_gts ----------------
__global__ void tal_k3_topk(const float* __restrict__ alig,
                            const float* __restrict__ anc,
                            const float* __restrict__ gt_bboxes,
                            const float* __restrict__ mask_gt,
                            int* __restrict__ tki, int* __restrict__ tkc) {
  const int g = blockIdx.x, b = blockIdx.y, tid = threadIdx.x;
  const int gi = b * NMAX_N + g;
  __shared__ float row[A_N];
  __shared__ float rv[256];
  __shared__ int   ri[256];
  __shared__ float box[4];
  __shared__ int   validS;
  if (tid == 0) validS = (mask_gt[gi] > 0.f) ? 1 : 0;
  if (tid < 4)  box[tid] = gt_bboxes[gi * 4 + tid];
  __syncthreads();
  if (!validS) { if (tid == 0) tkc[gi] = 0; return; }
  const float x1 = box[0], y1 = box[1], x2 = box[2], y2 = box[3];
  const float* arow = alig + (size_t)gi * A_N;
  for (int a = tid; a < A_N; a += 256) {
    float ax = anc[2 * a], ay = anc[2 * a + 1];
    float dmin = fminf(fminf(ax - x1, ay - y1), fminf(x2 - ax, y2 - ay));
    row[a] = (dmin > GEPS) ? arow[a] : 0.0f;   // metrics = align * mask_in_gts
  }
  __syncthreads();
  for (int t = 0; t < TOPK_N; ++t) {
    float bvv = -2.0f; int bii = A_N;
    for (int a = tid; a < A_N; a += 256) {
      float v = row[a];
      if (v > bvv) { bvv = v; bii = a; }       // ascending scan -> lowest idx on ties
    }
    rv[tid] = bvv; ri[tid] = bii;
    __syncthreads();
    for (int s = 128; s > 0; s >>= 1) {
      if (tid < s) {
        float v2 = rv[tid + s]; int i2 = ri[tid + s];
        if (v2 > rv[tid] || (v2 == rv[tid] && i2 < ri[tid])) { rv[tid] = v2; ri[tid] = i2; }
      }
      __syncthreads();
    }
    if (tid == 0) { tki[gi * TOPK_N + t] = ri[0]; row[ri[0]] = -1.0f; }
    __syncthreads();
  }
  if (tid == 0) tkc[gi] = TOPK_N;
}

// ---------------- K4: per-anchor column resolution + per-gt maxes ----------------
__global__ void tal_k4_resolve(const float* __restrict__ anc,
                               const float* __restrict__ gt_bboxes,
                               const float* __restrict__ ov,
                               const float* __restrict__ alig,
                               const int*   __restrict__ tki,
                               const int*   __restrict__ tkc,
                               float* __restrict__ aval,
                               int*   __restrict__ tgt,
                               int*   __restrict__ fg,
                               float* __restrict__ pal,
                               float* __restrict__ pov) {
  const int b = blockIdx.y, tid = threadIdx.x;
  const int a = blockIdx.x * 256 + tid;
  __shared__ __align__(16) float4 boxS[NMAX_N];
  __shared__ __align__(16) int listS[NMAX_N * TOPK_N];   // 832 ints = 208 x b128
  __shared__ __align__(16) int cntS[NMAX_N];             // 64 ints  = 16 x b128
  // async-stage all per-batch metadata (~4.6 KB) into LDS
  if (tid < NMAX_N) {
    async_copy_b128(&boxS[tid],
                    reinterpret_cast<const float4*>(gt_bboxes) + b * NMAX_N + tid);
  }
  if (tid < (NMAX_N * TOPK_N) / 4) {
    async_copy_b128(&listS[tid * 4], tki + b * NMAX_N * TOPK_N + tid * 4);
  }
  if (tid < NMAX_N / 4) {
    async_copy_b128(&cntS[tid * 4], tkc + b * NMAX_N + tid * 4);
  }
  async_wait0();
  __syncthreads();
  if (a >= A_N) return;
  const float ax = anc[2 * a], ay = anc[2 * a + 1];
  unsigned long long pm = 0ull; int sum = 0;
  for (int g = 0; g < NMAX_N; ++g) {
    if (cntS[g] == 0) continue;
    const int* l = &listS[g * TOPK_N];
    bool hit = false;
    #pragma unroll
    for (int j = 0; j < TOPK_N; ++j) hit = hit || (l[j] == a);
    if (!hit) continue;
    float4 bx = boxS[g];
    float dmin = fminf(fminf(ax - bx.x, ay - bx.y), fminf(bx.z - ax, bx.w - ay));
    if (dmin > GEPS) { pm |= (1ull << g); ++sum; }
  }
  int tg = 0, f = 0;
  if (sum > 1) {                       // multi-assigned: argmax over all gts of overlap
    const float* ocol = ov + (size_t)b * NMAX_N * A_N + a;
    float bvv = -1e30f; int bii = 0;
    for (int g = 0; g < NMAX_N; ++g) {
      float v = ocol[(size_t)g * A_N];
      if (v > bvv) { bvv = v; bii = g; }   // strict > -> first max (jnp.argmax)
    }
    tg = bii; f = 1;
  } else if (sum == 1) {
    tg = __ffsll(pm) - 1; f = 1;
  }
  const size_t ia = (size_t)b * A_N + a;
  tgt[ia] = tg; fg[ia] = f;
  float av = 0.f;
  if (f) {
    const size_t ig = ((size_t)b * NMAX_N + tg) * A_N + a;
    av = alig[ig];
    float o = ov[ig];
    atomicMax((int*)&pal[b * NMAX_N + tg], __float_as_int(av));   // values >= 0
    atomicMax((int*)&pov[b * NMAX_N + tg], __float_as_int(o));
  }
  aval[ia] = av;
}

// ---------------- K5: normalization + all five outputs ----------------
__global__ void tal_k5_out(const int*   __restrict__ gt_labels,
                           const float* __restrict__ gt_bboxes,
                           const float* __restrict__ aval,
                           const int*   __restrict__ tgt,
                           const int*   __restrict__ fg,
                           const float* __restrict__ pal,
                           const float* __restrict__ pov,
                           float* __restrict__ out_labels,
                           float* __restrict__ out_boxes,
                           float* __restrict__ out_scores,
                           float* __restrict__ out_fg,
                           float* __restrict__ out_idx) {
  const int b = blockIdx.y, tid = threadIdx.x;
  const int aBase = blockIdx.x * 256;
  const int a = aBase + tid;
  __shared__ int   labL[256];
  __shared__ float nrmL[256];
  __shared__ int   fgL[256];
  int lab = 0, f = 0; float nrm = 0.f;
  if (a < A_N) {
    const size_t ia = (size_t)b * A_N + a;
    const int t = tgt[ia]; f = fg[ia];
    const int gi = b * NMAX_N + t;
    lab = max(gt_labels[gi], 0);
    nrm = f ? aval[ia] * pov[gi] / (pal[gi] + GEPS) : 0.f;
    out_labels[ia] = (float)lab;
    out_fg[ia]     = f ? 1.f : 0.f;
    out_idx[ia]    = (float)t;
    reinterpret_cast<float4*>(out_boxes)[ia] =
        reinterpret_cast<const float4*>(gt_bboxes)[gi];
  }
  labL[tid] = lab; nrmL[tid] = nrm; fgL[tid] = f;
  __syncthreads();
  // coalesced one-hot score writes: block covers 256 anchors x 80 classes
  const size_t sbase = ((size_t)b * A_N + aBase) * NC_N;
  const int lim = (A_N - aBase) * NC_N;
  for (int i = 0; i < NC_N; ++i) {
    int e = i * 256 + tid;
    if (e < lim) {
      int al = e / NC_N, c = e % NC_N;
      float v = (fgL[al] && c == labL[al]) ? nrmL[al] : 0.f;
      out_scores[sbase + e] = v;
    }
  }
}

// ---------------- launcher ----------------
extern "C" void kernel_launch(void* const* d_in, const int* in_sizes, int n_in,
                              void* d_out, int out_size, void* d_ws, size_t ws_size,
                              hipStream_t stream) {
  (void)in_sizes; (void)n_in; (void)out_size; (void)ws_size;
  const float* pd_scores = (const float*)d_in[0];
  const float* pd_bboxes = (const float*)d_in[1];
  const float* anc       = (const float*)d_in[2];
  const int*   gt_labels = (const int*)d_in[3];
  const float* gt_bboxes = (const float*)d_in[4];
  const float* mask_gt   = (const float*)d_in[5];

  // workspace carving (~70 MB: two (16,64,8400) f32 maps + small buffers)
  char* base = (char*)d_ws;
  size_t off = 0;
  auto carve = [&](size_t bytes) -> void* {
    void* p = base + off;
    off = (off + bytes + 15) & ~(size_t)15;
    return p;
  };
  float* ov   = (float*)carve((size_t)BS_N * NMAX_N * A_N * 4);
  float* alig = (float*)carve((size_t)BS_N * NMAX_N * A_N * 4);
  int*   tki  = (int*)  carve((size_t)BS_N * NMAX_N * TOPK_N * 4);
  int*   tkc  = (int*)  carve((size_t)BS_N * NMAX_N * 4);
  float* pal  = (float*)carve((size_t)BS_N * NMAX_N * 4);
  float* pov  = (float*)carve((size_t)BS_N * NMAX_N * 4);
  float* aval = (float*)carve((size_t)BS_N * A_N * 4);
  int*   tgt  = (int*)  carve((size_t)BS_N * A_N * 4);
  int*   fgb  = (int*)  carve((size_t)BS_N * A_N * 4);

  // output layout: labels | bboxes | scores | fg | gt_idx (all as float)
  float* out_labels = (float*)d_out;
  float* out_boxes  = out_labels + (size_t)BS_N * A_N;
  float* out_scores = out_boxes  + (size_t)BS_N * A_N * 4;
  float* out_fg     = out_scores + (size_t)BS_N * A_N * NC_N;
  float* out_idx    = out_fg     + (size_t)BS_N * A_N;

  tal_k0_init   <<<dim3(1),            dim3(256), 0, stream>>>(pal, pov);
  tal_k1_ciou   <<<dim3(ATILES, BS_N), dim3(256), 0, stream>>>(pd_bboxes, gt_bboxes, ov);
  tal_k2_wmma   <<<dim3(WTILES, BS_N), dim3(32),  0, stream>>>(pd_scores, gt_labels, ov, alig);
  tal_k3_topk   <<<dim3(NMAX_N, BS_N), dim3(256), 0, stream>>>(alig, anc, gt_bboxes, mask_gt, tki, tkc);
  tal_k4_resolve<<<dim3(ATILES, BS_N), dim3(256), 0, stream>>>(anc, gt_bboxes, ov, alig, tki, tkc,
                                                               aval, tgt, fgb, pal, pov);
  tal_k5_out    <<<dim3(ATILES, BS_N), dim3(256), 0, stream>>>(gt_labels, gt_bboxes, aval, tgt, fgb,
                                                               pal, pov, out_labels, out_boxes,
                                                               out_scores, out_fg, out_idx);
}
